// AevaLinearAttention_33414845563490
// MI455X (gfx1250) — matrix operand order repaired and checked by reading, not compiled, not run on hardware
//
#include <hip/hip_runtime.h>
#include <hip/hip_fp16.h>

typedef __attribute__((ext_vector_type(16))) _Float16 v16h;
typedef __attribute__((ext_vector_type(8)))  float    v8f;

#define BB   2
#define SSQ  4096
#define HH   16
#define DDIM 128
#define HID  2048
#define CCH  128
#define NCH  32
#define MMR  (BB*SSQ)   // 8192 rows

union FA { v16h v; _Float16 e[16]; };
union AC { v8f  v; float    e[8];  };

__device__ inline v8f wmma16(v16h a, v16h b, v8f c) {
  // v_wmma_f32_16x16x32_f16
  return __builtin_amdgcn_wmma_f32_16x16x32_f16(false, a, false, b, (short)0, c,
                                                false, false);
}

// ---- CDNA5 async global->LDS (ASYNCcnt) --------------------------------
__device__ inline void async_b128(void* lds, const void* g) {
  asm volatile("global_load_async_to_lds_b128 %0, %1, off"
               :: "v"((unsigned)(uintptr_t)lds),
                  "v"((unsigned long long)(uintptr_t)g)
               : "memory");
}
__device__ inline void wait_async0() {
  asm volatile("s_wait_asynccnt 0" ::: "memory");
}

// A fragment: 16x32 f16, row-major source, leading dim ld.
__device__ inline FA load_a16x32(const _Float16* p, int ld, int lane) {
  FA f;
  int m = lane & 15, hi = lane >> 4;
  const _Float16* row = p + m * ld;
#pragma unroll
  for (int j = 0; j < 8; ++j) {
    int k = ((j >> 2) << 4) + (hi << 3) + ((j & 3) << 1);
    f.e[2 * j]     = row[k];
    f.e[2 * j + 1] = row[k + 1];
  }
  return f;
}

// B fragment: 32x16 f16. Lane L holds column N=L%16; 16 halves = K in table order.
__device__ inline FA load_b32x16(const _Float16* p, int ld, int lane) {
  FA f;
  int n = lane & 15, hi = lane >> 4;
#pragma unroll
  for (int i = 0; i < 16; ++i) {
    int k = ((i >> 3) << 4) + (hi << 3) + (i & 7);
    f.e[i] = p[k * ld + n];
  }
  return f;
}

__device__ inline void store_acc_f16(_Float16* p, int ld, int lane, const AC& a) {
  int nn = lane & 15, mb = (lane >> 4) << 3;
#pragma unroll
  for (int j = 0; j < 8; ++j) p[(mb + j) * ld + nn] = (_Float16)a.e[j];
}

// ---------------------------------------------------------------------------
__global__ void cvt_f32_f16(const float* __restrict__ in,
                            _Float16* __restrict__ out, int n) {
  int i = blockIdx.x * blockDim.x + threadIdx.x;
  int stride = gridDim.x * blockDim.x;
  for (; i < n; i += stride) out[i] = (_Float16)in[i];
}

// ---------------------------------------------------------------------------
// 128x128-tile GEMM, C = act(A @ B + bias), double-buffered async K-slabs.
#define LDA 72
#define LDB 136
#define ASLAB (128 * LDA)
#define BSLAB (64 * LDB)

__device__ inline void gemm_stage(const _Float16* __restrict__ A,
                                  const _Float16* __restrict__ Bw,
                                  _Float16* sAbuf, _Float16* sBbuf,
                                  int tid, int m0, int n0, int K, int N, int kt) {
  int r = tid >> 1, hs = tid & 1;
  const _Float16* ga = A + (size_t)(m0 + r) * K + kt + hs * 32;
  _Float16* la = sAbuf + r * LDA + hs * 32;
#pragma unroll
  for (int u = 0; u < 4; ++u) async_b128(la + u * 8, ga + u * 8);
  int kr = tid >> 2, q = tid & 3;
  const _Float16* gb = Bw + (size_t)(kt + kr) * N + n0 + q * 32;
  _Float16* lb = sBbuf + kr * LDB + q * 32;
#pragma unroll
  for (int u = 0; u < 4; ++u) async_b128(lb + u * 8, gb + u * 8);
}

template <int MODE>  // 0: f16 out. 1: f16 out + elu+1. 2: f32 out.
__global__ __launch_bounds__(256) void gemm_wmma(
    const _Float16* __restrict__ A, const _Float16* __restrict__ Bw,
    const float* __restrict__ bias, void* __restrict__ Cout,
    int M, int N, int K) {
  (void)M;
  extern __shared__ char gsmem[];
  _Float16* sA = (_Float16*)gsmem;
  _Float16* sB = sA + 2 * ASLAB;
  int tid = threadIdx.x, wave = tid >> 5, lane = tid & 31;
  int m0 = blockIdx.y * 128, n0 = blockIdx.x * 128;
  int wm = wave & 3, wn = wave >> 2;
  int rm = wm * 32, cn = wn * 64;

  AC acc[2][4];
#pragma unroll
  for (int a = 0; a < 2; ++a)
#pragma unroll
    for (int b = 0; b < 4; ++b)
#pragma unroll
      for (int j = 0; j < 8; ++j) acc[a][b].e[j] = 0.f;

  gemm_stage(A, Bw, sA, sB, tid, m0, n0, K, N, 0);

  for (int kt = 0, it = 0; kt < K; kt += 64, ++it) {
    int buf = it & 1;
    wait_async0();
    __syncthreads();
    if (kt + 64 < K)
      gemm_stage(A, Bw, sA + (buf ^ 1) * ASLAB, sB + (buf ^ 1) * BSLAB,
                 tid, m0, n0, K, N, kt + 64);
    const _Float16* cA = sA + buf * ASLAB;
    const _Float16* cB = sB + buf * BSLAB;
#pragma unroll
    for (int ks = 0; ks < 2; ++ks) {
      int kb = ks * 32;
      FA af0 = load_a16x32(cA + (rm + 0) * LDA + kb, LDA, lane);
      FA af1 = load_a16x32(cA + (rm + 16) * LDA + kb, LDA, lane);
#pragma unroll
      for (int tn = 0; tn < 4; ++tn) {
        FA bf = load_b32x16(cB + kb * LDB + cn + tn * 16, LDB, lane);
        acc[0][tn].v = wmma16(af0.v, bf.v, acc[0][tn].v);
        acc[1][tn].v = wmma16(af1.v, bf.v, acc[1][tn].v);
      }
    }
  }

  int nn = lane & 15, mb = (lane >> 4) << 3;
#pragma unroll
  for (int tn = 0; tn < 4; ++tn) {
    int gn = n0 + cn + tn * 16 + nn;
    float bv = bias[gn];
#pragma unroll
    for (int tm = 0; tm < 2; ++tm)
#pragma unroll
      for (int j = 0; j < 8; ++j) {
        int gm = m0 + rm + tm * 16 + mb + j;
        float v = acc[tm][tn].e[j] + bv;
        if (MODE == 1) v = v > 0.f ? v + 1.f : __expf(v);
        if (MODE == 2)
          ((float*)Cout)[(size_t)gm * N + gn] = v;
        else
          ((_Float16*)Cout)[(size_t)gm * N + gn] = (_Float16)v;
      }
  }
}

// ---------------------------------------------------------------------------
// Pass 1: per-chunk KV state. 1024 WGs, one per (b,h,chunk).
// kv[n] = kf^T @ v (f32 to global), ksum[n] = colsum(kf).
__global__ __launch_bounds__(256) void kv_chunk(
    const _Float16* __restrict__ kg, const _Float16* __restrict__ vg,
    float* __restrict__ kvg, float* __restrict__ ksg) {
  extern __shared__ char smem[];
  const int LD = 136;
  _Float16* sK  = (_Float16*)smem;
  _Float16* sV  = sK + 128 * LD;
  _Float16* sKT = sV + 128 * LD;

  int id = blockIdx.x;
  int n = id % NCH, bh = id / NCH;
  int b = bh / HH, h = bh % HH;
  int tid = threadIdx.x, wave = tid >> 5, lane = tid & 31;
  int wm = wave & 3, wn = wave >> 2;
  int nn = lane & 15, mb = (lane >> 4) << 3;
  int c = tid >> 1, hs = tid & 1;

  {  // async-stage kf/v chunk tiles
    size_t row = (size_t)(b * SSQ + n * CCH + c) * HID + (size_t)h * DDIM + hs * 64;
    _Float16* lk = sK + c * LD + hs * 64;
    _Float16* lv = sV + c * LD + hs * 64;
#pragma unroll
    for (int u = 0; u < 8; ++u) {
      async_b128(lk + u * 8, kg + row + u * 8);
      async_b128(lv + u * 8, vg + row + u * 8);
    }
  }
  wait_async0();
  __syncthreads();
  {  // transpose kf -> sKT
    const _Float16* src = sK + c * LD + hs * 64;
#pragma unroll
    for (int u = 0; u < 8; ++u) {
      uint4 pk = *(const uint4*)(src + u * 8);
      const _Float16* e = (const _Float16*)&pk;
#pragma unroll
      for (int i = 0; i < 8; ++i)
        sKT[(hs * 64 + u * 8 + i) * LD + c] = e[i];
    }
  }
  __syncthreads();

  float* kvdst = kvg + (size_t)id * (DDIM * DDIM);
#pragma unroll
  for (int t = 0; t < 8; ++t) {
    int tr = wm * 2 + (t >> 2), tc = wn * 4 + (t & 3);  // d-block, e-block
    AC a;
#pragma unroll
    for (int j = 0; j < 8; ++j) a.e[j] = 0.f;
#pragma unroll
    for (int kk = 0; kk < 4; ++kk) {
      FA af = load_a16x32(sKT + tr * 16 * LD + kk * 32, LD, lane);
      FA bf = load_b32x16(sV + kk * 32 * LD + tc * 16, LD, lane);
      a.v = wmma16(af.v, bf.v, a.v);
    }
#pragma unroll
    for (int j = 0; j < 8; ++j)
      kvdst[(size_t)(tr * 16 + mb + j) * DDIM + tc * 16 + nn] = a.e[j];
  }
  if (tid < 128) {
    float s = 0.f;
    const _Float16* ktrow = sKT + tid * LD;
#pragma unroll 8
    for (int c2 = 0; c2 < 128; ++c2) s += (float)ktrow[c2];
    ksg[(size_t)id * DDIM + tid] = s;
  }
}

// ---------------------------------------------------------------------------
// Pass 2: exclusive prefix scan over chunks. 32 WGs, one per (b,h).
// Each thread carries 64 f32 state elements in registers.
__global__ __launch_bounds__(256) void kv_scan(
    const float* __restrict__ kvg, const float* __restrict__ ksg,
    _Float16* __restrict__ kvex, float* __restrict__ ksex) {
  int bh = blockIdx.x;
  int tid = threadIdx.x;
  float S[64];
#pragma unroll
  for (int i = 0; i < 64; ++i) S[i] = 0.f;
  float sk = 0.f;
  for (int n = 0; n < NCH; ++n) {
    size_t base = ((size_t)bh * NCH + n) * (DDIM * DDIM);
#pragma unroll
    for (int i = 0; i < 64; ++i) {
      size_t idx = base + (size_t)i * 256 + tid;  // coalesced
      kvex[idx] = (_Float16)S[i];
      S[i] += kvg[idx];
    }
    if (tid < 128) {
      size_t kb = ((size_t)bh * NCH + n) * DDIM + tid;
      ksex[kb] = sk;
      sk += ksg[kb];
    }
  }
}

// ---------------------------------------------------------------------------
// Pass 3: stateless attention output. 1024 WGs, one per (b,h,chunk).
// out = (qf @ kv_ex + tril(qf kf^T) @ v) / (qf.ksum_ex + rowsum(A) + eps)
__global__ __launch_bounds__(256) void attn_out(
    const _Float16* __restrict__ qg, const _Float16* __restrict__ kg,
    const _Float16* __restrict__ vg, const _Float16* __restrict__ kvex,
    const float* __restrict__ ksex, _Float16* __restrict__ og) {
  extern __shared__ char smem[];
  const int LD = 136;
  _Float16* sQ  = (_Float16*)smem;        // qf chunk [128][LD]
  _Float16* sK  = sQ + 128 * LD;          // kf chunk
  _Float16* sV  = sK + 128 * LD;          // v chunk
  _Float16* sKT = sV + 128 * LD;          // kf^T
  _Float16* sA  = sKT + 128 * LD;         // masked scores
  _Float16* sS  = sA + 128 * LD;          // kv_ex (f16) [d][e], ld = 136
  float* zIn = (float*)(sS + 128 * LD);
  float* zTr = zIn + 128;
  float* sKs = zTr + 128;                 // ksum_ex [128]

  int id = blockIdx.x;
  int n = id % NCH, bh = id / NCH;
  int b = bh / HH, h = bh % HH;
  int tid = threadIdx.x, wave = tid >> 5, lane = tid & 31;
  int wm = wave & 3, wn = wave >> 2;
  int nn = lane & 15, mb = (lane >> 4) << 3;
  int c = tid >> 1, hs = tid & 1;
  const size_t colbase = (size_t)h * DDIM;

  {  // async-stage qf/kf/v and kv_ex
    size_t row = (size_t)(b * SSQ + n * CCH + c) * HID + colbase + hs * 64;
    _Float16* lq = sQ + c * LD + hs * 64;
    _Float16* lk = sK + c * LD + hs * 64;
    _Float16* lv = sV + c * LD + hs * 64;
    const _Float16* gs = kvex + (size_t)id * (DDIM * DDIM) + c * DDIM + hs * 64;
    _Float16* ls = sS + c * LD + hs * 64;
#pragma unroll
    for (int u = 0; u < 8; ++u) {
      async_b128(lq + u * 8, qg + row + u * 8);
      async_b128(lk + u * 8, kg + row + u * 8);
      async_b128(lv + u * 8, vg + row + u * 8);
      async_b128(ls + u * 8, gs + u * 8);
    }
  }
  if (tid < 128) sKs[tid] = ksex[(size_t)id * DDIM + tid];
  wait_async0();
  __syncthreads();

  {  // transpose kf -> sKT
    const _Float16* src = sK + c * LD + hs * 64;
#pragma unroll
    for (int u = 0; u < 8; ++u) {
      uint4 pk = *(const uint4*)(src + u * 8);
      const _Float16* e = (const _Float16*)&pk;
#pragma unroll
      for (int i = 0; i < 8; ++i)
        sKT[(hs * 64 + u * 8 + i) * LD + c] = e[i];
    }
  }
  __syncthreads();

  // step 1: A = tril(qf @ kf^T)
#pragma unroll
  for (int t = 0; t < 8; ++t) {
    int tr = wm * 2 + (t >> 2), tc = wn * 4 + (t & 3);
    AC a;
#pragma unroll
    for (int j = 0; j < 8; ++j) a.e[j] = 0.f;
#pragma unroll
    for (int kk = 0; kk < 4; ++kk) {
      FA af = load_a16x32(sQ + tr * 16 * LD + kk * 32, LD, lane);
      FA bf = load_b32x16(sKT + kk * 32 * LD + tc * 16, LD, lane);
      a.v = wmma16(af.v, bf.v, a.v);
    }
#pragma unroll
    for (int j = 0; j < 8; ++j) {
      int rr = tr * 16 + mb + j, cc2 = tc * 16 + nn;
      if (cc2 > rr) a.e[j] = 0.f;
    }
    store_acc_f16(sA + tr * 16 * LD + tc * 16, LD, lane, a);
  }
  __syncthreads();

  // normalizers
  if (tid < 128) {
    float zt = 0.f, zi = 0.f;
    const _Float16* arow = sA + tid * LD;
    const _Float16* qrow = sQ + tid * LD;
#pragma unroll 8
    for (int k2 = 0; k2 < 128; ++k2) {
      zt += (float)arow[k2];
      zi += (float)qrow[k2] * sKs[k2];
    }
    zTr[tid] = zt;
    zIn[tid] = zi;
  }
  __syncthreads();

  // step 2: out = qf @ kv_ex + A @ v, normalize, store
#pragma unroll
  for (int t = 0; t < 8; ++t) {
    int tr = wm * 2 + (t >> 2), tc = wn * 4 + (t & 3);
    AC o;
#pragma unroll
    for (int j = 0; j < 8; ++j) o.e[j] = 0.f;
#pragma unroll
    for (int kk = 0; kk < 4; ++kk) {
      FA af = load_a16x32(sQ + tr * 16 * LD + kk * 32, LD, lane);
      FA bf = load_b32x16(sS + kk * 32 * LD + tc * 16, LD, lane);
      o.v = wmma16(af.v, bf.v, o.v);
    }
#pragma unroll
    for (int kk = 0; kk < 4; ++kk) {
      FA af = load_a16x32(sA + tr * 16 * LD + kk * 32, LD, lane);
      FA bf = load_b32x16(sV + kk * 32 * LD + tc * 16, LD, lane);
      o.v = wmma16(af.v, bf.v, o.v);
    }
#pragma unroll
    for (int j = 0; j < 8; ++j) {
      int cr = tr * 16 + mb + j;
      float z = zIn[cr] + zTr[cr] + 1e-6f;
      og[(size_t)(b * SSQ + n * CCH + cr) * HID + colbase + tc * 16 + nn] =
          (_Float16)(o.e[j] / z);
    }
  }
}

// ---------------------------------------------------------------------------
extern "C" void kernel_launch(void* const* d_in, const int* in_sizes, int n_in,
                              void* d_out, int out_size, void* d_ws, size_t ws_size,
                              hipStream_t stream) {
  (void)in_sizes; (void)n_in; (void)out_size; (void)ws_size;
  const float* x  = (const float*)d_in[0];
  const float* Wq = (const float*)d_in[1];
  const float* bq = (const float*)d_in[2];
  const float* Wk = (const float*)d_in[3];
  const float* bk = (const float*)d_in[4];
  const float* Wv = (const float*)d_in[5];
  const float* bv = (const float*)d_in[6];
  const float* Wo = (const float*)d_in[7];
  const float* bo = (const float*)d_in[8];
  float* out = (float*)d_out;

  char* ws = (char*)d_ws;
  size_t off = 0;
  _Float16* xh  = (_Float16*)(ws + off); off += (size_t)MMR * HID * 2;
  _Float16* wqh = (_Float16*)(ws + off); off += (size_t)HID * HID * 2;
  _Float16* wkh = (_Float16*)(ws + off); off += (size_t)HID * HID * 2;
  _Float16* wvh = (_Float16*)(ws + off); off += (size_t)HID * HID * 2;
  _Float16* woh = (_Float16*)(ws + off); off += (size_t)HID * HID * 2;
  _Float16* qf  = (_Float16*)(ws + off); off += (size_t)MMR * HID * 2;
  _Float16* kf  = (_Float16*)(ws + off); off += (size_t)MMR * HID * 2;
  _Float16* vf  = (_Float16*)(ws + off); off += (size_t)MMR * HID * 2;
  _Float16* ah  = (_Float16*)(ws + off); off += (size_t)MMR * HID * 2;
  float*    kvg = (float*)(ws + off);    off += (size_t)BB*HH*NCH*DDIM*DDIM*4;
  float*    ksg = (float*)(ws + off);    off += (size_t)BB*HH*NCH*DDIM*4;
  _Float16* kvex= (_Float16*)(ws + off); off += (size_t)BB*HH*NCH*DDIM*DDIM*2;
  float*    ksex= (float*)(ws + off);    off += (size_t)BB*HH*NCH*DDIM*4;

  cvt_f32_f16<<<2048, 256, 0, stream>>>(x, xh, MMR * HID);
  cvt_f32_f16<<<1024, 256, 0, stream>>>(Wq, wqh, HID * HID);
  cvt_f32_f16<<<1024, 256, 0, stream>>>(Wk, wkh, HID * HID);
  cvt_f32_f16<<<1024, 256, 0, stream>>>(Wv, wvh, HID * HID);
  cvt_f32_f16<<<1024, 256, 0, stream>>>(Wo, woh, HID * HID);

  size_t gsm = (size_t)2 * (ASLAB + BSLAB) * 2;  // ~70 KB
  (void)hipFuncSetAttribute((const void*)gemm_wmma<0>,
                            hipFuncAttributeMaxDynamicSharedMemorySize, (int)gsm);
  (void)hipFuncSetAttribute((const void*)gemm_wmma<1>,
                            hipFuncAttributeMaxDynamicSharedMemorySize, (int)gsm);
  (void)hipFuncSetAttribute((const void*)gemm_wmma<2>,
                            hipFuncAttributeMaxDynamicSharedMemorySize, (int)gsm);

  dim3 grid(HID / 128, MMR / 128), blk(256);
  gemm_wmma<1><<<grid, blk, gsm, stream>>>(xh, wqh, bq, qf, MMR, HID, HID);
  gemm_wmma<1><<<grid, blk, gsm, stream>>>(xh, wkh, bk, kf, MMR, HID, HID);
  gemm_wmma<0><<<grid, blk, gsm, stream>>>(xh, wvh, bv, vf, MMR, HID, HID);

  size_t sm1 = (size_t)3 * 128 * 136 * 2;                   // ~104 KB
  (void)hipFuncSetAttribute((const void*)kv_chunk,
                            hipFuncAttributeMaxDynamicSharedMemorySize, (int)sm1);
  kv_chunk<<<BB * HH * NCH, 256, sm1, stream>>>(kf, vf, kvg, ksg);

  kv_scan<<<BB * HH, 256, 0, stream>>>(kvg, ksg, kvex, ksex);

  size_t sm2 = (size_t)6 * 128 * 136 * 2 + 3 * 128 * 4;     // ~210 KB
  (void)hipFuncSetAttribute((const void*)attn_out,
                            hipFuncAttributeMaxDynamicSharedMemorySize, (int)sm2);
  attn_out<<<BB * HH * NCH, 256, sm2, stream>>>(qf, kf, vf, kvex, ksex, ah);

  gemm_wmma<2><<<grid, blk, gsm, stream>>>(ah, woh, bo, out, MMR, HID, HID);
}
